// MultiHeadSelfAttention_20624432955881
// MI455X (gfx1250) — matrix operand order repaired and checked by reading, not compiled, not run on hardware
//
#include <hip/hip_runtime.h>

typedef __attribute__((ext_vector_type(16))) __bf16 v16bf;
typedef __attribute__((ext_vector_type(8)))  __bf16 v8bf;
typedef __attribute__((ext_vector_type(8)))  float  v8f;

#define D_MODEL 1024
#define S_LEN   2048
#define NHEAD   16
#define DEPTH   64
#define MROWS   4096   // B * S

__device__ __forceinline__ v8f wmma_bf16(v16bf a, v16bf b, v8f c) {
  return __builtin_amdgcn_wmma_f32_16x16x32_bf16(false, a, false, b, (short)0, c,
                                                 false, false);
}

__device__ __forceinline__ v16bf cat16(v8bf lo, v8bf hi) {
  v16bf r;
#pragma unroll
  for (int i = 0; i < 8; ++i) { r[i] = lo[i]; r[i + 8] = hi[i]; }
  return r;
}

__device__ __forceinline__ v8f vzero8() {
  v8f r;
#pragma unroll
  for (int i = 0; i < 8; ++i) r[i] = 0.0f;
  return r;
}

// Async DMA of 16 bytes/lane from global to LDS (gfx1250 ASYNCcnt path).
__device__ __forceinline__ void async_copy_b128(const void* gsrc, void* lds_dst) {
  unsigned ldsa = (unsigned)(unsigned long long)lds_dst;  // LDS aperture: addr[31:0]
  unsigned long long ga = (unsigned long long)gsrc;
  asm volatile("global_load_async_to_lds_b128 %0, %1, off"
               :: "v"(ldsa), "v"(ga) : "memory");
}

__global__ void cvt_f32_bf16(const float* __restrict__ in, __bf16* __restrict__ out, int n) {
  int i = blockIdx.x * blockDim.x + threadIdx.x;
  if (i < n) out[i] = (__bf16)in[i];
}

// y = x @ W.T + b ; split heads. mode 0 -> Q[B,H,S,64], 1 -> K[B,H,S,64], 2 -> V^T[B,H,64,S]
__global__ __launch_bounds__(256) void qkv_gemm(
    const __bf16* __restrict__ xb,
    const __bf16* __restrict__ wqb, const __bf16* __restrict__ wkb,
    const __bf16* __restrict__ wvb,
    const float* __restrict__ bq, const float* __restrict__ bk,
    const float* __restrict__ bv,
    __bf16* __restrict__ Q, __bf16* __restrict__ K, __bf16* __restrict__ Vt) {
  const int lane = threadIdx.x & 31, wave = threadIdx.x >> 5;
  const int lmod = lane & 15, lhalf = lane >> 4;
  const int m0 = blockIdx.x * 128 + wave * 16;
  const int n0 = blockIdx.y * 64;
  const int mode = blockIdx.z;
  const __bf16* W   = (mode == 0) ? wqb : (mode == 1) ? wkb : wvb;
  const float* bias = (mode == 0) ? bq  : (mode == 1) ? bk  : bv;

  v8f acc[4];
#pragma unroll
  for (int t = 0; t < 4; ++t) acc[t] = vzero8();

  const int arow = m0 + lmod;
  for (int k0 = 0; k0 < D_MODEL; k0 += 32) {
    const __bf16* ap = xb + (size_t)arow * D_MODEL + k0 + lhalf * 16;
    v16bf a = cat16(*(const v8bf*)ap, *(const v8bf*)(ap + 8));
#pragma unroll
    for (int t = 0; t < 4; ++t) {
      const __bf16* bp = W + (size_t)(n0 + t * 16 + lmod) * D_MODEL + k0 + lhalf * 16;
      acc[t] = wmma_bf16(a, *(const v16bf*)bp, acc[t]);
    }
  }

#pragma unroll
  for (int t = 0; t < 4; ++t) {
#pragma unroll
    for (int v = 0; v < 8; ++v) {
      const int n = n0 + t * 16 + lmod;
      const int m = m0 + v + 8 * lhalf;
      const float val = acc[t][v] + bias[n];
      const int b = m >> 11, s = m & (S_LEN - 1);
      const int h = n >> 6,  d = n & 63;
      const __bf16 bv16 = (__bf16)val;
      if (mode == 2)
        Vt[(((size_t)(b * NHEAD + h)) * DEPTH + d) * S_LEN + s] = bv16;
      else if (mode == 1)
        K[(((size_t)(b * NHEAD + h)) * S_LEN + s) * DEPTH + d] = bv16;
      else
        Q[(((size_t)(b * NHEAD + h)) * S_LEN + s) * DEPTH + d] = bv16;
    }
  }
}

// Flash attention: 4 waves/block share one (b,h); K/V chunks staged to LDS via
// async global->LDS DMA, double-buffered; online softmax; P re-laid-out via LDS.
__global__ __launch_bounds__(128) void flash_attn(
    const __bf16* __restrict__ Q, const __bf16* __restrict__ K,
    const __bf16* __restrict__ Vt, __bf16* __restrict__ Aout) {
  __shared__ __align__(32) __bf16 kbuf[2][32 * DEPTH];  // [key 0..31][d 0..63]
  __shared__ __align__(32) __bf16 vbuf[2][DEPTH * 32];  // [d 0..63][key 0..31]
  __shared__ __align__(32) __bf16 plds[4][16 * 32];

  const int tid = threadIdx.x;
  const int lane = tid & 31, wave = tid >> 5;
  const int lmod = lane & 15, lhalf = lane >> 4;
  const int qt = blockIdx.x * 4 + wave;  // 0..127
  const int bh = blockIdx.y;             // 0..31
  const int b = bh >> 4, h = bh & 15;

  const __bf16* Qp = Q  + (size_t)bh * S_LEN * DEPTH;
  const __bf16* Kp = K  + (size_t)bh * S_LEN * DEPTH;
  const __bf16* Vp = Vt + (size_t)bh * DEPTH * S_LEN;

  // stage chunk c (keys [c*32, c*32+32)) into buffer c&1: 4 async b128 per thread
  auto stage = [&](int c) {
    const int cb = c & 1;
    const size_t kc = (size_t)c * 32;
    // K chunk: 32 rows x 64 bf16 = 4KB, contiguous in memory
#pragma unroll
    for (int i = 0; i < 2; ++i) {
      const int off = i * 2048 + tid * 16;  // bytes
      async_copy_b128((const char*)(Kp + kc * DEPTH) + off,
                      (char*)&kbuf[cb][0] + off);
    }
    // V chunk: 64 rows (d) x 32 bf16 (s), row stride S_LEN
#pragma unroll
    for (int i = 0; i < 2; ++i) {
      const int off = i * 2048 + tid * 16;  // bytes within 4KB tile
      const int row = off >> 6, col = off & 63;
      async_copy_b128((const char*)(Vp + (size_t)row * S_LEN + kc) + col,
                      (char*)&vbuf[cb][0] + off);
    }
  };

  const int qrow = qt * 16 + lmod;
  v16bf qa[2];
#pragma unroll
  for (int kk = 0; kk < 2; ++kk) {
    const __bf16* ap = Qp + (size_t)qrow * DEPTH + kk * 32 + lhalf * 16;
    qa[kk] = cat16(*(const v8bf*)ap, *(const v8bf*)(ap + 8));
  }

  v8f o[4];
#pragma unroll
  for (int t = 0; t < 4; ++t) o[t] = vzero8();
  float mrow[8], lrow[8];
#pragma unroll
  for (int v = 0; v < 8; ++v) { mrow[v] = -1.0e30f; lrow[v] = 0.0f; }

  const float scale = 0.125f;  // 1/sqrt(64)
  const int NC = S_LEN / 32;   // 64 chunks

  stage(0);
  for (int c = 0; c < NC; ++c) {
    const int cb = c & 1;
    if (c + 1 < NC) {
      stage(c + 1);
      asm volatile("s_wait_asynccnt 0x4" ::: "memory");  // chunk c landed
    } else {
      asm volatile("s_wait_asynccnt 0x0" ::: "memory");
    }
    __syncthreads();  // all waves' DMA portions visible

    // S = Q * K^T for 32 keys: two 16x16 accumulators, depth 64 -> 2 k-steps
    v8f s0 = vzero8(), s1 = vzero8();
#pragma unroll
    for (int kk = 0; kk < 2; ++kk) {
      const __bf16* k0p = &kbuf[cb][(size_t)lmod * DEPTH + kk * 32 + lhalf * 16];
      const __bf16* k1p = &kbuf[cb][(size_t)(16 + lmod) * DEPTH + kk * 32 + lhalf * 16];
      s0 = wmma_bf16(qa[kk], *(const v16bf*)k0p, s0);
      s1 = wmma_bf16(qa[kk], *(const v16bf*)k1p, s1);
    }

    float alpha[8];
#pragma unroll
    for (int v = 0; v < 8; ++v) {
      float rm = fmaxf(s0[v], s1[v]) * scale;
#pragma unroll
      for (int off = 1; off < 16; off <<= 1)
        rm = fmaxf(rm, __shfl_xor(rm, off, 32));
      const float mnew = fmaxf(mrow[v], rm);
      alpha[v] = __expf(mrow[v] - mnew);
      mrow[v] = mnew;
      const float p0 = __expf(s0[v] * scale - mnew);
      const float p1 = __expf(s1[v] * scale - mnew);
      float rs = p0 + p1;
#pragma unroll
      for (int off = 1; off < 16; off <<= 1)
        rs += __shfl_xor(rs, off, 32);
      lrow[v] = lrow[v] * alpha[v] + rs;
      const int m = v + 8 * lhalf;
      plds[wave][m * 32 + lmod]      = (__bf16)p0;
      plds[wave][m * 32 + 16 + lmod] = (__bf16)p1;
    }

#pragma unroll
    for (int t = 0; t < 4; ++t)
#pragma unroll
      for (int v = 0; v < 8; ++v) o[t][v] *= alpha[v];

    asm volatile("s_wait_dscnt 0x0" ::: "memory");
    const __bf16* pp = &plds[wave][lmod * 32];
    v16bf pa = cat16(*(const v8bf*)(pp + lhalf * 16),
                     *(const v8bf*)(pp + lhalf * 16 + 8));
#pragma unroll
    for (int t = 0; t < 4; ++t) {
      const __bf16* vp = &vbuf[cb][(size_t)(t * 16 + lmod) * 32 + lhalf * 16];
      o[t] = wmma_bf16(pa, *(const v16bf*)vp, o[t]);
    }
    __syncthreads();  // protect buffer cb before stage(c+2) overwrites it
  }

#pragma unroll
  for (int t = 0; t < 4; ++t)
#pragma unroll
    for (int v = 0; v < 8; ++v) {
      const int m = v + 8 * lhalf;
      const int srow = qt * 16 + m;
      const int d = t * 16 + lmod;
      const float val = o[t][v] / lrow[v];
      Aout[((size_t)(b * S_LEN + srow)) * D_MODEL + h * DEPTH + d] = (__bf16)val;
    }
}

// out = attn @ Wd.T + bd  (f32 output)
__global__ __launch_bounds__(256) void out_gemm(
    const __bf16* __restrict__ ab, const __bf16* __restrict__ wdb,
    const float* __restrict__ bd, float* __restrict__ out) {
  const int lane = threadIdx.x & 31, wave = threadIdx.x >> 5;
  const int lmod = lane & 15, lhalf = lane >> 4;
  const int m0 = blockIdx.x * 128 + wave * 16;
  const int n0 = blockIdx.y * 64;

  v8f acc[4];
#pragma unroll
  for (int t = 0; t < 4; ++t) acc[t] = vzero8();

  const int arow = m0 + lmod;
  for (int k0 = 0; k0 < D_MODEL; k0 += 32) {
    const __bf16* ap = ab + (size_t)arow * D_MODEL + k0 + lhalf * 16;
    v16bf a = cat16(*(const v8bf*)ap, *(const v8bf*)(ap + 8));
#pragma unroll
    for (int t = 0; t < 4; ++t) {
      const __bf16* bp = wdb + (size_t)(n0 + t * 16 + lmod) * D_MODEL + k0 + lhalf * 16;
      acc[t] = wmma_bf16(a, *(const v16bf*)bp, acc[t]);
    }
  }

#pragma unroll
  for (int t = 0; t < 4; ++t)
#pragma unroll
    for (int v = 0; v < 8; ++v) {
      const int n = n0 + t * 16 + lmod;
      const int m = m0 + v + 8 * lhalf;
      out[(size_t)m * D_MODEL + n] = acc[t][v] + bd[n];
    }
}

extern "C" void kernel_launch(void* const* d_in, const int* in_sizes, int n_in,
                              void* d_out, int out_size, void* d_ws, size_t ws_size,
                              hipStream_t stream) {
  const float* x  = (const float*)d_in[0];
  const float* wq = (const float*)d_in[1];
  const float* bq = (const float*)d_in[2];
  const float* wk = (const float*)d_in[3];
  const float* bk = (const float*)d_in[4];
  const float* wv = (const float*)d_in[5];
  const float* bv = (const float*)d_in[6];
  const float* wd = (const float*)d_in[7];
  const float* bd = (const float*)d_in[8];
  float* out = (float*)d_out;

  char* w = (char*)d_ws;
  const size_t MB = (size_t)1 << 20;
  __bf16* xb  = (__bf16*)(w + 0 * MB);   // 8 MB   x as bf16 [4096,1024]
  __bf16* wqb = (__bf16*)(w + 8 * MB);   // 2 MB
  __bf16* wkb = (__bf16*)(w + 10 * MB);  // 2 MB
  __bf16* wvb = (__bf16*)(w + 12 * MB);  // 2 MB
  __bf16* wdb = (__bf16*)(w + 14 * MB);  // 2 MB
  __bf16* Qb  = (__bf16*)(w + 16 * MB);  // 8 MB  [B,H,S,64]
  __bf16* Kb  = (__bf16*)(w + 24 * MB);  // 8 MB  [B,H,S,64]
  __bf16* Vtb = (__bf16*)(w + 32 * MB);  // 8 MB  [B,H,64,S]
  __bf16* Ab  = (__bf16*)(w + 40 * MB);  // 8 MB  attn output [B,S,H*64]

  const int nx = MROWS * D_MODEL;
  const int nw = D_MODEL * D_MODEL;
  cvt_f32_bf16<<<(nx + 255) / 256, 256, 0, stream>>>(x, xb, nx);
  cvt_f32_bf16<<<(nw + 255) / 256, 256, 0, stream>>>(wq, wqb, nw);
  cvt_f32_bf16<<<(nw + 255) / 256, 256, 0, stream>>>(wk, wkb, nw);
  cvt_f32_bf16<<<(nw + 255) / 256, 256, 0, stream>>>(wv, wvb, nw);
  cvt_f32_bf16<<<(nw + 255) / 256, 256, 0, stream>>>(wd, wdb, nw);

  qkv_gemm<<<dim3(32, 16, 3), 256, 0, stream>>>(xb, wqb, wkb, wvb, bq, bk, bv,
                                                Qb, Kb, Vtb);
  flash_attn<<<dim3(32, 32), 128, 0, stream>>>(Qb, Kb, Vtb, Ab);
  out_gemm<<<dim3(32, 16), 256, 0, stream>>>(Ab, wdb, bd, out);
}
